// TunableLutLinear_55551107006781
// MI455X (gfx1250) — matrix operand order repaired and checked by reading, not compiled, not run on hardware
//
#include <hip/hip_runtime.h>

// ---------------------------------------------------------------------------
// TunableLutLinear on MI455X (gfx1250, wave32, WMMA + TDM)
//  Stage 0: x f32 -> bf16 (xb in d_ws)            [memory-bound, one pass]
//  Stage 1: softmax(logits) x luts -> W bf16      [memory-bound, one pass]
//  Stage 2: out = x @ W^T + bias; tiles staged by TENSOR_LOAD_TO_LDS (TDM),
//           math via v_wmma_f32_16x16x32_bf16
// ---------------------------------------------------------------------------

#define OUT_DIM   4096
#define IN_DIM    4096
#define NUM_LUT   2
#define LUT_SIZE  16
#define VEC       8
#define GROUPS    (IN_DIM / VEC)   // 512
#define TOKENS    8192

typedef __attribute__((ext_vector_type(8)))  float        v8f;
typedef __attribute__((ext_vector_type(16))) __bf16       v16bf;
typedef __attribute__((ext_vector_type(4)))  unsigned int v4u;
typedef __attribute__((ext_vector_type(8)))  int          v8i;
typedef __attribute__((ext_vector_type(4)))  int          v4i;

union FragBF {       // 32 bytes = 8 VGPRs, assembled from two 16B LDS loads
    v16bf v;
    uint4 q[2];
};

__device__ __forceinline__ unsigned short f2bf(float f) {
    unsigned int u = __builtin_bit_cast(unsigned int, f);
    u += 0x7FFFu + ((u >> 16) & 1u);        // round-to-nearest-even
    return (unsigned short)(u >> 16);
}

__device__ __forceinline__ unsigned int pk2bf(float lo, float hi) {
    return (unsigned int)f2bf(lo) | ((unsigned int)f2bf(hi) << 16);
}

// ---------------------------------------------------------------------------
// Stage 0: x (f32) -> xb (bf16), 8 elements per thread, 16B stores.
// ---------------------------------------------------------------------------
__global__ void __launch_bounds__(256)
cvt_x_kernel(const float* __restrict__ x, unsigned short* __restrict__ xb)
{
    const size_t i = ((size_t)blockIdx.x * 256 + threadIdx.x) * 8;
    const float4 a = *(const float4*)(x + i);
    const float4 b = *(const float4*)(x + i + 4);
    uint4 st = make_uint4(pk2bf(a.x, a.y), pk2bf(a.z, a.w),
                          pk2bf(b.x, b.y), pk2bf(b.z, b.w));
    *(uint4*)(xb + i) = st;
}

// ---------------------------------------------------------------------------
// Stage 1: build bf16 weight.  grid = (OUT/256, GROUPS), block = 256.
// ---------------------------------------------------------------------------
__global__ void __launch_bounds__(256)
build_weight_kernel(const float* __restrict__ logits,
                    const float* __restrict__ luts,
                    unsigned short* __restrict__ W)
{
    __shared__ float slut[NUM_LUT][LUT_SIZE][VEC];

    const int g   = blockIdx.y;
    const int tid = threadIdx.x;

    {
        const int n = tid >> 7;          // 0..1
        const int r = tid & 127;         // 0..127
        slut[n][r >> 3][r & 7] =
            luts[((size_t)n * GROUPS + g) * (LUT_SIZE * VEC) + r];
    }
    __syncthreads();

    const int o = blockIdx.x * 256 + tid;

    float acc[VEC];
#pragma unroll
    for (int v = 0; v < VEC; ++v) acc[v] = 0.0f;

#pragma unroll
    for (int n = 0; n < NUM_LUT; ++n) {
        const float* lg = logits +
            (((size_t)n * OUT_DIM + o) * GROUPS + g) * LUT_SIZE;

        float l[LUT_SIZE];
        const float4* lg4 = (const float4*)lg;
#pragma unroll
        for (int q = 0; q < LUT_SIZE / 4; ++q) {
            float4 t = lg4[q];
            l[q * 4 + 0] = t.x; l[q * 4 + 1] = t.y;
            l[q * 4 + 2] = t.z; l[q * 4 + 3] = t.w;
        }

        float m = l[0];
#pragma unroll
        for (int k = 1; k < LUT_SIZE; ++k) m = fmaxf(m, l[k]);
        float s = 0.0f;
#pragma unroll
        for (int k = 0; k < LUT_SIZE; ++k) { l[k] = __expf(l[k] - m); s += l[k]; }
        const float inv = __frcp_rn(s);

#pragma unroll
        for (int k = 0; k < LUT_SIZE; ++k) {
            const float p = l[k] * inv;
#pragma unroll
            for (int v = 0; v < VEC; ++v)
                acc[v] = fmaf(p, slut[n][k][v], acc[v]);
        }
    }

    uint4 st = make_uint4(pk2bf(acc[0], acc[1]), pk2bf(acc[2], acc[3]),
                          pk2bf(acc[4], acc[5]), pk2bf(acc[6], acc[7]));
    *(uint4*)(&W[(size_t)o * IN_DIM + (size_t)g * VEC]) = st;
}

// ---------------------------------------------------------------------------
// Stage 2: GEMM  out[T,O] = xb[T,I] * W[O,I]^T + bias   (bf16 in, f32 acc)
// Block tile 128(M) x 256(N) x 32(K); 8 waves (2x4); wave tile 64x64
//   = 16 v_wmma per K-step from 16 ds_load_b128.
// Tiles staged by TDM (tensor_load_to_lds), double-buffered, 1 barrier/iter.
// unroll(1) + launch_bounds(256,4): keep wave inside 256 directly-addressable
// VGPRs (no s_set_vgpr_msb / cross-bank v_mov traffic), 4 waves/SIMD.
// ---------------------------------------------------------------------------
#define BM 128
#define BN 256
#define BK 32
#define NKITER (IN_DIM / BK)   // 128
#define LDS_STRIDE 40          // 32 bf16 + 16B pad (TDM pad: 16DW interval, 4DW pad)

// Issue one 2-D TDM tile load: `rows` x 32 bf16 elements from a row-major
// [rows x IN_DIM] bf16 tensor slice starting at gptr, into LDS rows of
// LDS_STRIDE bf16 (64B data + 16B pad per row).
__device__ __forceinline__ void tdm_load_tile(unsigned lds_addr,
                                              const unsigned short* gptr,
                                              int rows)
{
    const unsigned long long ga = (unsigned long long)(uintptr_t)gptr;

    v4u g0;
    g0[0] = 1u;                                   // count=1, user descriptor
    g0[1] = lds_addr;                             // LDS byte address
    g0[2] = (unsigned int)ga;                     // global_addr[31:0]
    g0[3] = (unsigned int)(ga >> 32) | (2u << 30);// global_addr[56:32] | type=2

    v8i g1;
    g1[0] = 0x06D10000;          // data_size=2B | pad_enable | ivl=16DW | amt=4DW
    g1[1] = (int)((IN_DIM & 0xFFFF) << 16);       // tensor_dim0[15:0]=4096
    g1[2] = (int)((rows & 0xFFFF) << 16);         // dim0 hi=0 | tensor_dim1 lo
    g1[3] = (int)(32u << 16);                     // dim1 hi=0 | tile_dim0=32
    g1[4] = rows & 0xFFFF;                        // tile_dim1=rows | tile_dim2=0
    g1[5] = IN_DIM;                               // tensor_dim0_stride lo
    g1[6] = 0;                                    // stride0 hi | stride1 lo
    g1[7] = 0;                                    // stride1 hi

    const v4i gz4 = {0, 0, 0, 0};                 // groups 2/3: 2-D tile
    const v8i gz8 = {0, 0, 0, 0, 0, 0, 0, 0};     // extension group (unused)
    __builtin_amdgcn_tensor_load_to_lds(g0, g1, gz4, gz4, gz8, 0);
}

__global__ void __launch_bounds__(256, 4)
lut_gemm_kernel(const unsigned short* __restrict__ xb,
                const unsigned short* __restrict__ W,
                const float* __restrict__ bias,
                float* __restrict__ out)
{
    __shared__ unsigned short sA[2][BM][LDS_STRIDE];   // 2 x 10 KB
    __shared__ unsigned short sB[2][BN][LDS_STRIDE];   // 2 x 20 KB

    const int tid   = threadIdx.x;
    const int lane  = tid & 31;
    const int wave  = tid >> 5;        // 0..7
    const int waveM = wave >> 2;       // 0..1  -> 64 rows each
    const int waveN = wave & 3;        // 0..3  -> 64 cols each

    const int m0 = blockIdx.x * BM;    // token base
    const int n0 = blockIdx.y * BN;    // out-feature base

    const int lm = lane & 15;          // M/N index within a 16x16 tile
    const int kh = lane >> 4;          // half-wave selector

    v8f acc[4][4];
    const v8f vzero = {0.f, 0.f, 0.f, 0.f, 0.f, 0.f, 0.f, 0.f};
#pragma unroll
    for (int mt = 0; mt < 4; ++mt)
#pragma unroll
        for (int nt = 0; nt < 4; ++nt) acc[mt][nt] = vzero;

    const unsigned short* aG = xb + (size_t)m0 * IN_DIM;
    const unsigned short* bG = W  + (size_t)n0 * IN_DIM;

    const unsigned ldsA[2] = { (unsigned)(uintptr_t)&sA[0][0][0],
                               (unsigned)(uintptr_t)&sA[1][0][0] };
    const unsigned ldsB[2] = { (unsigned)(uintptr_t)&sB[0][0][0],
                               (unsigned)(uintptr_t)&sB[1][0][0] };

    // prologue: DMA first K-slice into buffer 0
    if (wave == 0) {
        tdm_load_tile(ldsA[0], aG, BM);
        tdm_load_tile(ldsB[0], bG, BN);
        __builtin_amdgcn_s_wait_tensorcnt(0);
    }
    __syncthreads();

#pragma unroll 1
    for (int kci = 0; kci < NKITER; ++kci) {
        const int cur = kci & 1;

        // async-stage next K-slice into the other buffer (TDM, TENSORcnt)
        if (wave == 0 && kci + 1 < NKITER) {
            const int kc = (kci + 1) * BK;
            tdm_load_tile(ldsA[cur ^ 1], aG + kc, BM);
            tdm_load_tile(ldsB[cur ^ 1], bG + kc, BN);
        }

        const unsigned short (*A)[LDS_STRIDE] = sA[cur];
        const unsigned short (*B)[LDS_STRIDE] = sB[cur];

        FragBF a[4];
#pragma unroll
        for (int mt = 0; mt < 4; ++mt) {
            const int row = waveM * 64 + mt * 16 + lm;
            a[mt].q[0] = *(const uint4*)(&A[row][kh * 8]);       // K=kh*8..+7
            a[mt].q[1] = *(const uint4*)(&A[row][16 + kh * 8]);  // K=16+kh*8..+7
        }
#pragma unroll
        for (int nt = 0; nt < 4; ++nt) {
            const int col = waveN * 64 + nt * 16 + lm;
            FragBF b;
            b.q[0] = *(const uint4*)(&B[col][kh * 8]);           // K=kh*8..+7
            b.q[1] = *(const uint4*)(&B[col][16 + kh * 8]);      // K=16+kh*8..+7
#pragma unroll
            for (int mt = 0; mt < 4; ++mt)
                acc[mt][nt] = __builtin_amdgcn_wmma_f32_16x16x32_bf16(
                    false, a[mt].v, false, b.v,
                    (short)0, acc[mt][nt], false, false);
        }

        // publisher wave drains the DMA before the whole group flips buffers
        if (wave == 0 && kci + 1 < NKITER)
            __builtin_amdgcn_s_wait_tensorcnt(0);
        __syncthreads();
    }

    // ---- epilogue: + bias, store ----
#pragma unroll
    for (int nt = 0; nt < 4; ++nt) {
        const int col = n0 + waveN * 64 + nt * 16 + lm;
        const float bv = bias[col];
#pragma unroll
        for (int mt = 0; mt < 4; ++mt) {
            const int rbase = m0 + waveM * 64 + mt * 16 + kh * 8;
#pragma unroll
            for (int r = 0; r < 8; ++r)
                out[(size_t)(rbase + r) * OUT_DIM + col] = acc[mt][nt][r] + bv;
        }
    }
}

// ---------------------------------------------------------------------------
extern "C" void kernel_launch(void* const* d_in, const int* in_sizes, int n_in,
                              void* d_out, int out_size, void* d_ws, size_t ws_size,
                              hipStream_t stream) {
    const float* x      = (const float*)d_in[0];  // [8192,4096]
    const float* logits = (const float*)d_in[1];  // [2,4096,512,16]
    const float* luts   = (const float*)d_in[2];  // [2,512,16,8]
    const float* bias   = (const float*)d_in[3];  // [4096]
    float*       out    = (float*)d_out;          // [8192,4096]

    unsigned short* xb = (unsigned short*)d_ws;                       // 64 MB
    unsigned short* W  = xb + (size_t)TOKENS * IN_DIM;                // 32 MB

    // Stage 0: x -> bf16 (removes cvt VALU from GEMM hot loop, halves A traffic)
    cvt_x_kernel<<<(TOKENS * IN_DIM / 8) / 256, 256, 0, stream>>>(x, xb);

    // Stage 1: synthesize weight (memory-bound, ~256 MB logits stream)
    build_weight_kernel<<<dim3(OUT_DIM / 256, GROUPS), 256, 0, stream>>>(
        logits, luts, W);

    // Stage 2: WMMA GEMM (274 GFLOP, bf16 in / f32 acc, TDM-staged tiles)
    lut_gemm_kernel<<<dim3(TOKENS / BM, OUT_DIM / BN), 256, 0, stream>>>(
        xb, W, bias, out);
}
